// Generator_79688823210567
// MI455X (gfx1250) — matrix operand order repaired and checked by reading, not compiled, not run on hardware
//
#include <hip/hip_runtime.h>
#include <hip/hip_bf16.h>

// Problem constants (match reference)
#define B_    512
#define T_    1024
#define H_    64
#define MH_   128
#define DOUT_ 16

typedef __bf16 bf16;
typedef __attribute__((ext_vector_type(16))) __bf16 v16bf;  // 8 VGPRs (WMMA A/B operand)
typedef __attribute__((ext_vector_type(8)))  float  v8f;    // 8 VGPRs (WMMA C/D operand)

// ---------------------------------------------------------------------------
// WMMA wrapper: D = A(16x32 bf16) x B(32x16 bf16) + C(16x16 f32)
// ---------------------------------------------------------------------------
static __device__ __forceinline__ v8f wmma_bf16(v16bf a, v16bf b, v8f c) {
  return __builtin_amdgcn_wmma_f32_16x16x32_bf16(
      /*neg_a=*/false, a, /*neg_b=*/false, b,
      /*c_mod=*/(short)0, c, /*reuse_a=*/false, /*reuse_b=*/false);
}

union AView { v16bf v; bf16 h[16]; uint4 q[2]; };

// A fragment (16x32 bf16 tile) from LDS, row-major activation buffer.
// Lane layout: lanes 0-15 row M=lane holds K = kc+[0..7] and kc+[16..23];
//              lanes 16-31 row M=lane-16 holds K = kc+[8..15] and kc+[24..31].
static __device__ __forceinline__ v16bf lds_atile(const bf16* act, int ldw, int kc, int lane) {
  const int r   = lane & 15;
  const int klo = kc + ((lane >> 4) << 3);
  const bf16* p = act + r * ldw + klo;
  AView u;
  u.q[0] = *reinterpret_cast<const uint4*>(p);        // 8 bf16 (16 B)
  u.q[1] = *reinterpret_cast<const uint4*>(p + 16);   // 8 bf16 (16 B)
  return u.v;
}

// A fragment from global bf16 row-major matrix (ld elements per row).
static __device__ __forceinline__ v16bf gbl_atile_bf16(const bf16* X, int ld, int row0, int kc, int lane) {
  const int r   = row0 + (lane & 15);
  const int klo = kc + ((lane >> 4) << 3);
  const bf16* p = X + (size_t)r * ld + klo;
  AView u;
  u.q[0] = *reinterpret_cast<const uint4*>(p);
  u.q[1] = *reinterpret_cast<const uint4*>(p + 16);
  return u.v;
}

// A fragment from global f32 row-major matrix, converted to bf16.
static __device__ __forceinline__ v16bf gbl_atile_f32(const float* X, int ld, int row0, int kc, int lane) {
  const int r   = row0 + (lane & 15);
  const int klo = kc + ((lane >> 4) << 3);
  const float* p = X + (size_t)r * ld + klo;
  AView u;
#pragma unroll
  for (int i = 0; i < 8; ++i) {
    u.h[i]     = (bf16)p[i];
    u.h[8 + i] = (bf16)p[16 + i];
  }
  return u.v;
}

// B fragment (32x16 bf16 tile) from global f32 weight W stored (K,N) row-major.
// Lane layout: lanes 0-15 col N=n0+lane holds K = k0+[0..7] and k0+[16..23];
//              lanes 16-31 col N=n0+lane-16 holds K = k0+[8..15] and k0+[24..31].
// One-time cost (weights are register-resident afterwards).
static __device__ __forceinline__ v16bf gbl_btile(const float* W, int ldn, int k0, int n0, int lane) {
  const int n  = n0 + (lane & 15);
  const int kk = k0 + ((lane >> 4) << 3);
  AView u;
#pragma unroll
  for (int i = 0; i < 16; ++i) u.h[i] = (bf16)W[(size_t)(kk + i + ((i >> 3) << 3)) * ldn + n];
  return u.v;
}

// C tile (16x16 f32) -> bias (+ optional ReLU) -> bf16 LDS activation buffer.
// C/D layout: element (vgpr v, lane l): M = v + 8*(l>>4), N = l&15.
static __device__ __forceinline__ void ctile_to_act(v8f c, bf16* act, int ldw, int n0, int lane,
                                                    float bias, bool relu) {
  const int n  = n0 + (lane & 15);
  const int mb = (lane >> 4) << 3;
#pragma unroll
  for (int v = 0; v < 8; ++v) {
    float x = c[v] + bias;
    if (relu) x = fmaxf(x, 0.0f);
    act[(mb + v) * ldw + n] = (bf16)x;
  }
}

// ---------------------------------------------------------------------------
// Kernel 1: x0 = init MLP(init_noise)   64 -> 128 -> 128 -> 64
// One block per 16 batch rows. 8 waves: wave w owns cols [16w,16w+16) of L1/L2;
// waves 0-3 own cols [16w,16w+16) of L3.
// ---------------------------------------------------------------------------
__global__ void __launch_bounds__(256) sde_init_kernel(
    const float* __restrict__ noise,
    const float* __restrict__ W0, const float* __restrict__ b0,
    const float* __restrict__ W1, const float* __restrict__ b1,
    const float* __restrict__ W2, const float* __restrict__ b2,
    float* __restrict__ x0out) {
  __shared__ alignas(16) bf16 act1[16 * 136];
  __shared__ alignas(16) bf16 act2[16 * 136];

  const int tid  = threadIdx.x;
  const int lane = tid & 31;
  const int wid  = tid >> 5;
  const int rb   = blockIdx.x * 16;

  v16bf B1t[2] = {}, B2t[4] = {}, B3t[4] = {};
#pragma unroll
  for (int kc = 0; kc < 2; ++kc) B1t[kc] = gbl_btile(W0, MH_, kc * 32, wid * 16, lane);
#pragma unroll
  for (int kc = 0; kc < 4; ++kc) B2t[kc] = gbl_btile(W1, MH_, kc * 32, wid * 16, lane);
  if (wid < 4) {
#pragma unroll
    for (int kc = 0; kc < 4; ++kc) B3t[kc] = gbl_btile(W2, H_, kc * 32, wid * 16, lane);
  }
  const float b0n = b0[wid * 16 + (lane & 15)];
  const float b1n = b1[wid * 16 + (lane & 15)];
  const float b2n = (wid < 4) ? b2[wid * 16 + (lane & 15)] : 0.0f;

  v8f c = {};
#pragma unroll
  for (int kc = 0; kc < 2; ++kc) {
    v16bf a = gbl_atile_f32(noise, 64, rb, kc * 32, lane);
    c = wmma_bf16(a, B1t[kc], c);
  }
  ctile_to_act(c, act1, 136, wid * 16, lane, b0n, true);
  __syncthreads();

  v8f d = {};
#pragma unroll
  for (int kc = 0; kc < 4; ++kc) {
    v16bf a = lds_atile(act1, 136, kc * 32, lane);
    d = wmma_bf16(a, B2t[kc], d);
  }
  ctile_to_act(d, act2, 136, wid * 16, lane, b1n, true);
  __syncthreads();

  if (wid < 4) {
    v8f e = {};
#pragma unroll
    for (int kc = 0; kc < 4; ++kc) {
      v16bf a = lds_atile(act2, 136, kc * 32, lane);
      e = wmma_bf16(a, B3t[kc], e);
    }
    const int n  = wid * 16 + (lane & 15);
    const int mb = (lane >> 4) << 3;
#pragma unroll
    for (int v = 0; v < 8; ++v)
      x0out[(size_t)(rb + mb + v) * H_ + n] = e[v] + b2n;
  }
}

// ---------------------------------------------------------------------------
// Kernel 2: reversible-Heun scan, 1023 steps.
// Batch rows are independent: 32 blocks x 16 rows, no inter-block sync.
// 8 waves: waves 0-3 = drift MLP, waves 4-7 = diffusion MLP.
// Wave (mlp,q): cols [32q,32q+32) of L1/L2, cols [16q,16q+16) of L3.
// Weights register-resident (bf16 fragments); activations via LDS; carry in LDS.
// Time column folded into layer-1 bias: b0[n] + t*W0[0][n].   (dt == 1)
// ---------------------------------------------------------------------------
__global__ void __launch_bounds__(256) sde_scan_kernel(
    const float* __restrict__ dWg,
    const float* __restrict__ x0,
    const float* __restrict__ fW0, const float* __restrict__ fb0,
    const float* __restrict__ fW1, const float* __restrict__ fb1,
    const float* __restrict__ fW2, const float* __restrict__ fb2,
    const float* __restrict__ gW0, const float* __restrict__ gb0,
    const float* __restrict__ gW1, const float* __restrict__ gb1,
    const float* __restrict__ gW2, const float* __restrict__ gb2,
    bf16* __restrict__ xs) {
  __shared__ alignas(16) float sY[16 * 64];
  __shared__ alignas(16) float sYh[16 * 64];
  __shared__ alignas(16) float sF[16 * 64];
  __shared__ alignas(16) float sG[16 * 64];
  __shared__ alignas(16) float fnew[2][16 * 64];
  __shared__ alignas(16) bf16 actA[16 * 72];        // state in bf16 (A operand), K=64
  __shared__ alignas(16) bf16 act1[2][16 * 136];
  __shared__ alignas(16) bf16 act2[2][16 * 136];

  const int tid  = threadIdx.x;
  const int lane = tid & 31;
  const int wid  = tid >> 5;
  const int mlp  = wid >> 2;  // 0 = drift, 1 = diffusion
  const int q    = wid & 3;
  const int rb   = blockIdx.x * 16;

  const float* W0 = mlp ? gW0 : fW0;
  const float* b0 = mlp ? gb0 : fb0;
  const float* W1 = mlp ? gW1 : fW1;
  const float* b1 = mlp ? gb1 : fb1;
  const float* W2 = mlp ? gW2 : fW2;
  const float* b2 = mlp ? gb2 : fb2;

  // ---- load weight fragments into VGPRs (one time) ----
  v16bf B1t[2][2], B2t[2][4], B3t[4];
#pragma unroll
  for (int i = 0; i < 2; ++i)
#pragma unroll
    for (int kc = 0; kc < 2; ++kc)
      B1t[i][kc] = gbl_btile(W0 + MH_ /* skip t row */, MH_, kc * 32, q * 32 + i * 16, lane);
#pragma unroll
  for (int i = 0; i < 2; ++i)
#pragma unroll
    for (int kc = 0; kc < 4; ++kc)
      B2t[i][kc] = gbl_btile(W1, MH_, kc * 32, q * 32 + i * 16, lane);
#pragma unroll
  for (int kc = 0; kc < 4; ++kc)
    B3t[kc] = gbl_btile(W2, H_, kc * 32, q * 16, lane);

  float b0n[2], w0r0[2], b1n[2];
#pragma unroll
  for (int i = 0; i < 2; ++i) {
    const int n = q * 32 + i * 16 + (lane & 15);
    b0n[i]  = b0[n];
    w0r0[i] = W0[n];  // row 0 of W0 = time-input weights
    b1n[i]  = b1[n];
  }
  const float b2n = b2[q * 16 + (lane & 15)];

  // ---- one full f&g evaluation (both MLPs run concurrently across waves) ----
  auto fg_pass = [&](float tval) {
    __syncthreads();  // actA ready
    bf16* a1 = &act1[mlp][0];
    bf16* a2 = &act2[mlp][0];
    // Layer 1: K=64 (state only; t folded into bias)
    v8f c0 = {}, c1 = {};
#pragma unroll
    for (int kc = 0; kc < 2; ++kc) {
      v16bf a = lds_atile(actA, 72, kc * 32, lane);
      c0 = wmma_bf16(a, B1t[0][kc], c0);
      c1 = wmma_bf16(a, B1t[1][kc], c1);
    }
    ctile_to_act(c0, a1, 136, q * 32, lane, b0n[0] + tval * w0r0[0], true);
    ctile_to_act(c1, a1, 136, q * 32 + 16, lane, b0n[1] + tval * w0r0[1], true);
    __syncthreads();
    // Layer 2: K=128
    v8f d0 = {}, d1 = {};
#pragma unroll
    for (int kc = 0; kc < 4; ++kc) {
      v16bf a = lds_atile(a1, 136, kc * 32, lane);
      d0 = wmma_bf16(a, B2t[0][kc], d0);
      d1 = wmma_bf16(a, B2t[1][kc], d1);
    }
    ctile_to_act(d0, a2, 136, q * 32, lane, b1n[0], true);
    ctile_to_act(d1, a2, 136, q * 32 + 16, lane, b1n[1], true);
    __syncthreads();
    // Layer 3: K=128, N=64 (16 cols per wave), f32 result into fnew[mlp]
    v8f e0 = {};
#pragma unroll
    for (int kc = 0; kc < 4; ++kc) {
      v16bf a = lds_atile(a2, 136, kc * 32, lane);
      e0 = wmma_bf16(a, B3t[kc], e0);
    }
    {
      const int n  = q * 16 + (lane & 15);
      const int mb = (lane >> 4) << 3;
#pragma unroll
      for (int v = 0; v < 8; ++v) fnew[mlp][(mb + v) * 64 + n] = e0[v] + b2n;
    }
    __syncthreads();  // fnew ready
  };

  // ---- elementwise thread mapping: one float4 of the 16x64 state each ----
  const int r    = tid >> 4;
  const int h4   = (tid & 15) << 2;
  const int eoff = r * 64 + h4;
  const size_t xsrow = (size_t)(rb + r) * T_;

  // init carry: y = yh = x0; emit xs[:,0,:]
  {
    float4 x = *reinterpret_cast<const float4*>(x0 + (size_t)(rb + r) * H_ + h4);
    *reinterpret_cast<float4*>(&sY[eoff])  = x;
    *reinterpret_cast<float4*>(&sYh[eoff]) = x;
    union { bf16 h[4]; uint2 u; } pk;
    pk.h[0] = (bf16)x.x; pk.h[1] = (bf16)x.y; pk.h[2] = (bf16)x.z; pk.h[3] = (bf16)x.w;
    *reinterpret_cast<uint2*>(&actA[r * 72 + h4]) = pk.u;
    *reinterpret_cast<uint2*>(xs + (xsrow + 0) * H_ + h4) = pk.u;
  }
  fg_pass(0.0f);  // f0, g0
  {
    float4 f1 = *reinterpret_cast<const float4*>(&fnew[0][eoff]);
    float4 g1 = *reinterpret_cast<const float4*>(&fnew[1][eoff]);
    *reinterpret_cast<float4*>(&sF[eoff]) = f1;
    *reinterpret_cast<float4*>(&sG[eoff]) = g1;
  }

  const float dt = 1.0f;  // ts = arange(T) -> dt == 1, sqrt(dt) == 1
  for (int t1 = 1; t1 < T_; ++t1) {
    const float* dwp = dWg + ((size_t)(t1 - 1) * B_ + (rb + r)) * H_ + h4;
    float4 dw = *reinterpret_cast<const float4*>(dwp);
    if (t1 + 1 < T_) __builtin_prefetch(dwp + (size_t)B_ * H_, 0, 0);  // global_prefetch

    float4 y  = *reinterpret_cast<const float4*>(&sY[eoff]);
    float4 yh = *reinterpret_cast<const float4*>(&sYh[eoff]);
    float4 f  = *reinterpret_cast<const float4*>(&sF[eoff]);
    float4 g  = *reinterpret_cast<const float4*>(&sG[eoff]);

    float4 yh1;
    yh1.x = 2.0f * y.x - yh.x + f.x * dt + g.x * dw.x;
    yh1.y = 2.0f * y.y - yh.y + f.y * dt + g.y * dw.y;
    yh1.z = 2.0f * y.z - yh.z + f.z * dt + g.z * dw.z;
    yh1.w = 2.0f * y.w - yh.w + f.w * dt + g.w * dw.w;
    *reinterpret_cast<float4*>(&sYh[eoff]) = yh1;
    {
      union { bf16 h[4]; uint2 u; } pk;
      pk.h[0] = (bf16)yh1.x; pk.h[1] = (bf16)yh1.y; pk.h[2] = (bf16)yh1.z; pk.h[3] = (bf16)yh1.w;
      *reinterpret_cast<uint2*>(&actA[r * 72 + h4]) = pk.u;
    }

    fg_pass((float)t1);  // f1, g1 at (t1, yh1)

    float4 f1 = *reinterpret_cast<const float4*>(&fnew[0][eoff]);
    float4 g1 = *reinterpret_cast<const float4*>(&fnew[1][eoff]);
    float4 y1;
    y1.x = y.x + 0.5f * (f.x + f1.x) * dt + 0.5f * (g.x + g1.x) * dw.x;
    y1.y = y.y + 0.5f * (f.y + f1.y) * dt + 0.5f * (g.y + g1.y) * dw.y;
    y1.z = y.z + 0.5f * (f.z + f1.z) * dt + 0.5f * (g.z + g1.z) * dw.z;
    y1.w = y.w + 0.5f * (f.w + f1.w) * dt + 0.5f * (g.w + g1.w) * dw.w;
    *reinterpret_cast<float4*>(&sY[eoff]) = y1;
    *reinterpret_cast<float4*>(&sF[eoff]) = f1;
    *reinterpret_cast<float4*>(&sG[eoff]) = g1;
    {
      union { bf16 h[4]; uint2 u; } pk;
      pk.h[0] = (bf16)y1.x; pk.h[1] = (bf16)y1.y; pk.h[2] = (bf16)y1.z; pk.h[3] = (bf16)y1.w;
      *reinterpret_cast<uint2*>(xs + (xsrow + t1) * H_ + h4) = pk.u;
    }
    // no barrier needed: sY/sF/sG/sYh are thread-owned; actA/fnew are fenced inside fg_pass
  }
}

// ---------------------------------------------------------------------------
// Kernel 3: readout over all B*T rows.  65 -> 128 -> 128 -> 16
// Grid-stride over 16-row tiles; xs already bf16 in A-friendly row-major form.
// Per-row time feature (t % 24) folded as rank-1 bias in the L1 epilogue.
// ---------------------------------------------------------------------------
__global__ void __launch_bounds__(256) sde_readout_kernel(
    const bf16* __restrict__ xs,
    const float* __restrict__ W0, const float* __restrict__ b0,
    const float* __restrict__ W1, const float* __restrict__ b1,
    const float* __restrict__ W2, const float* __restrict__ b2,
    float* __restrict__ out) {
  __shared__ alignas(16) bf16 act1[16 * 136];
  __shared__ alignas(16) bf16 act2[16 * 136];

  const int tid  = threadIdx.x;
  const int lane = tid & 31;
  const int wid  = tid >> 5;

  v16bf B1t[2] = {}, B2t[4] = {}, B3t[4] = {};
#pragma unroll
  for (int kc = 0; kc < 2; ++kc) B1t[kc] = gbl_btile(W0 + MH_, MH_, kc * 32, wid * 16, lane);
#pragma unroll
  for (int kc = 0; kc < 4; ++kc) B2t[kc] = gbl_btile(W1, MH_, kc * 32, wid * 16, lane);
  if (wid == 0) {
#pragma unroll
    for (int kc = 0; kc < 4; ++kc) B3t[kc] = gbl_btile(W2, DOUT_, kc * 32, 0, lane);
  }
  const float b0n   = b0[wid * 16 + (lane & 15)];
  const float w0r0n = W0[wid * 16 + (lane & 15)];  // time-input weights (row 0)
  const float b1n   = b1[wid * 16 + (lane & 15)];
  const float b2n   = (wid == 0) ? b2[lane & 15] : 0.0f;

  const int NT = (B_ * T_) / 16;
  for (int tile = blockIdx.x; tile < NT; tile += gridDim.x) {
    const int rowbase = tile * 16;           // 16 consecutive (b,t) rows, same b
    const int tbase   = rowbase & (T_ - 1);

    // Layer 1 (K = 64 state cols; per-row t%24 applied in epilogue)
    v8f c = {};
#pragma unroll
    for (int kc = 0; kc < 2; ++kc) {
      v16bf a = gbl_atile_bf16(xs, H_, rowbase, kc * 32, lane);
      c = wmma_bf16(a, B1t[kc], c);
    }
    {
      const int n  = wid * 16 + (lane & 15);
      const int mb = (lane >> 4) << 3;
#pragma unroll
      for (int v = 0; v < 8; ++v) {
        const int m = mb + v;
        const float tm = (float)((tbase + m) % 24);
        float x = c[v] + b0n + tm * w0r0n;
        act1[m * 136 + n] = (bf16)fmaxf(x, 0.0f);
      }
    }
    __syncthreads();

    // Layer 2
    v8f d = {};
#pragma unroll
    for (int kc = 0; kc < 4; ++kc) {
      v16bf a = lds_atile(act1, 136, kc * 32, lane);
      d = wmma_bf16(a, B2t[kc], d);
    }
    ctile_to_act(d, act2, 136, wid * 16, lane, b1n, true);
    __syncthreads();

    // Layer 3 (N=16): wave 0 only
    if (wid == 0) {
      v8f e = {};
#pragma unroll
      for (int kc = 0; kc < 4; ++kc) {
        v16bf a = lds_atile(act2, 136, kc * 32, lane);
        e = wmma_bf16(a, B3t[kc], e);
      }
      const int n  = lane & 15;
      const int mb = (lane >> 4) << 3;
#pragma unroll
      for (int v = 0; v < 8; ++v)
        out[(size_t)(rowbase + mb + v) * DOUT_ + n] = e[v] + b2n;
    }
    __syncthreads();
  }
}

// ---------------------------------------------------------------------------
// Host launcher.
// d_in order (pytree-flattened setup_inputs):
//  0 init_noise, 1 dW,
//  2..7   init:    W0(64x128)  b0  W1(128x128) b1  W2(128x64)  b2
//  8..13  drift:   W0(65x128)  b0  W1(128x128) b1  W2(128x64)  b2
// 14..19  diff:    W0(65x128)  b0  W1(128x128) b1  W2(128x64)  b2
// 20..25  readout: W0(65x128)  b0  W1(128x128) b1  W2(128x16)  b2
// ws: [ xs bf16 (B*T*H) | x0 f32 (B*H) ]
// ---------------------------------------------------------------------------
extern "C" void kernel_launch(void* const* d_in, const int* in_sizes, int n_in,
                              void* d_out, int out_size, void* d_ws, size_t ws_size,
                              hipStream_t stream) {
  (void)in_sizes; (void)n_in; (void)out_size; (void)ws_size;

  const float* init_noise = (const float*)d_in[0];
  const float* dW         = (const float*)d_in[1];

  const float* iW0 = (const float*)d_in[2];
  const float* ib0 = (const float*)d_in[3];
  const float* iW1 = (const float*)d_in[4];
  const float* ib1 = (const float*)d_in[5];
  const float* iW2 = (const float*)d_in[6];
  const float* ib2 = (const float*)d_in[7];

  const float* fW0 = (const float*)d_in[8];
  const float* fb0 = (const float*)d_in[9];
  const float* fW1 = (const float*)d_in[10];
  const float* fb1 = (const float*)d_in[11];
  const float* fW2 = (const float*)d_in[12];
  const float* fb2 = (const float*)d_in[13];

  const float* gW0 = (const float*)d_in[14];
  const float* gb0 = (const float*)d_in[15];
  const float* gW1 = (const float*)d_in[16];
  const float* gb1 = (const float*)d_in[17];
  const float* gW2 = (const float*)d_in[18];
  const float* gb2 = (const float*)d_in[19];

  const float* rW0 = (const float*)d_in[20];
  const float* rb0 = (const float*)d_in[21];
  const float* rW1 = (const float*)d_in[22];
  const float* rb1 = (const float*)d_in[23];
  const float* rW2 = (const float*)d_in[24];
  const float* rb2 = (const float*)d_in[25];

  bf16*  xs = (bf16*)d_ws;
  float* x0 = (float*)((char*)d_ws + (size_t)B_ * T_ * H_ * sizeof(bf16));

  sde_init_kernel<<<B_ / 16, 256, 0, stream>>>(init_noise, iW0, ib0, iW1, ib1, iW2, ib2, x0);

  sde_scan_kernel<<<B_ / 16, 256, 0, stream>>>(dW, x0,
                                               fW0, fb0, fW1, fb1, fW2, fb2,
                                               gW0, gb0, gW1, gb1, gW2, gb2,
                                               xs);

  sde_readout_kernel<<<1024, 256, 0, stream>>>(xs, rW0, rb0, rW1, rb1, rW2, rb2,
                                               (float*)d_out);
}
